// CharacterModel_50869592654493
// MI455X (gfx1250) — compile-verified, matrix operand order checked by reading, not run on hardware
//
#include <hip/hip_runtime.h>
#include <hip/hip_bf16.h>

typedef __attribute__((ext_vector_type(16))) __bf16 bf16x16;
typedef __attribute__((ext_vector_type(8)))  __bf16 bf16x8;
typedef __attribute__((ext_vector_type(8)))  float  v8f;
typedef __attribute__((ext_vector_type(4)))  float  v4f;

#define NUM_WORDS 16384
#define MAX_WLEN  16
#define EMB       64
#define HID       128
#define WAVES_PER_BLOCK 4
#define WORDS_PER_BLOCK (WAVES_PER_BLOCK * 16)

__device__ __forceinline__ float sigf(float x) { return 1.0f / (1.0f + __expf(-x)); }
__device__ __forceinline__ float tanhf_fast(float x) {
    float e = __expf(2.0f * x);
    return (e - 1.0f) / (e + 1.0f);
}

__device__ __forceinline__ bf16x16 combine16(bf16x8 lo, bf16x8 hi) {
    bf16x16 r;
#pragma unroll
    for (int i = 0; i < 8; ++i) { r[i] = lo[i]; r[8 + i] = hi[i]; }
    return r;
}

// Load one lane's slice of a 32x16 bf16 B-tile (or 16x32 A-tile; identical
// per-lane K pattern) from a row-major bf16 row pointer.
// Lanes 0-15: halves 0..7 -> K = koff+0..7,   halves 8..15 -> K = koff+16..23
// Lanes 16-31: same +8.  hi8 = (lane>=16) ? 8 : 0.
__device__ __forceinline__ bf16x16 load_tile_bf16(const __bf16* row, int koff, int hi8) {
    bf16x8 lo = *(const bf16x8*)(row + koff + hi8);
    bf16x8 hi = *(const bf16x8*)(row + koff + 16 + hi8);
    return combine16(lo, hi);
}

// Same pattern but sourced from f32 data (embedding rows), converted to bf16.
__device__ __forceinline__ bf16x16 load_tile_f32(const float* p, int hi8) {
    v4f x0 = *(const v4f*)(p + hi8);
    v4f x1 = *(const v4f*)(p + hi8 + 4);
    v4f x2 = *(const v4f*)(p + hi8 + 16);
    v4f x3 = *(const v4f*)(p + hi8 + 20);
    bf16x16 r;
#pragma unroll
    for (int i = 0; i < 4; ++i) {
        r[i]      = (__bf16)x0[i];
        r[4 + i]  = (__bf16)x1[i];
        r[8 + i]  = (__bf16)x2[i];
        r[12 + i] = (__bf16)x3[i];
    }
    return r;
}

// ---------------------------------------------------------------------------
// Kernel 1: convert weights f32 -> bf16 in workspace, fold biases.
// ---------------------------------------------------------------------------
__global__ void convert_weights_kernel(const float* __restrict__ w_ih,
                                       const float* __restrict__ w_hh,
                                       const float* __restrict__ b_ih,
                                       const float* __restrict__ b_hh,
                                       __bf16* __restrict__ wih_b,
                                       __bf16* __restrict__ whh_b,
                                       float* __restrict__ bsum) {
    int i = blockIdx.x * blockDim.x + threadIdx.x;
    if (i < 512 * EMB) wih_b[i] = (__bf16)w_ih[i];
    if (i < 512 * HID) whh_b[i] = (__bf16)w_hh[i];
    if (i < 512)       bsum[i]  = b_ih[i] + b_hh[i];
}

// ---------------------------------------------------------------------------
// Kernel 2: fused LSTM + attention, one 16-word tile per wave32.
// Register-resident: only the cell state c (64 VGPRs, C/D layout).
// h lives in the per-wave LDS staging tile (doubles as the WMMA A-operand
// source); the attention result accumulates in LDS. All LDS traffic is
// same-wave and DS ops are in-order per wave -> no barriers, per-wave early
// exit on sorted word lengths is legal.
// ---------------------------------------------------------------------------
__global__ __launch_bounds__(128) void lstm_wmma_kernel(
    const int*    __restrict__ chars,          // [NUM_WORDS][MAX_WLEN]
    const int*    __restrict__ word_orig_idx,  // [NUM_WORDS]
    const int*    __restrict__ wordlens,       // [NUM_WORDS]
    const float*  __restrict__ emb,            // [VOCAB][EMB]
    const float*  __restrict__ attn,           // [HID]
    const float*  __restrict__ h0,             // [HID]
    const float*  __restrict__ c0,             // [HID]
    const __bf16* __restrict__ wih,            // [512][EMB] bf16
    const __bf16* __restrict__ whh,            // [512][HID] bf16
    const float*  __restrict__ bsum,           // [512]
    float*        __restrict__ out)            // [NUM_WORDS][HID]
{
    __shared__ __bf16 hstage[WAVES_PER_BLOCK][16][HID];  // 16 KB: h state / A relayout
    __shared__ float  resbuf[WAVES_PER_BLOCK][16][HID];  // 32 KB: attention accumulator

    const int lane = threadIdx.x & 31;
    const int wave = threadIdx.x >> 5;
    const int half = lane >> 4;          // 0: lanes 0-15, 1: lanes 16-31
    const int n    = lane & 15;          // N (column) index inside the tile
    const int hi8  = half ? 8 : 0;
    const int tile = blockIdx.x * WORDS_PER_BLOCK + wave * 16;

    // Pull weights toward the caches (gfx1250 global_prefetch path).
    __builtin_prefetch(wih + (threadIdx.x << 8), 0, 0);
    __builtin_prefetch(whh + (threadIdx.x << 9), 0, 0);

    // Word lengths for the two C/D-layout rows each lane owns per VGPR slot r.
    int wl[8];
#pragma unroll
    for (int r = 0; r < 8; ++r) wl[r] = wordlens[tile + r + 8 * half];

    // Register-resident cell state in C/D layout: element (M = r+8*half, N = n).
    v8f creg[8];
#pragma unroll
    for (int hn = 0; hn < 8; ++hn) {
        float cv = c0[hn * 16 + n];
        float hv = h0[hn * 16 + n];
#pragma unroll
        for (int r = 0; r < 8; ++r) {
            creg[hn][r] = cv;
            hstage[wave][r + 8 * half][hn * 16 + n] = (__bf16)hv;  // stage h0
            resbuf[wave][r + 8 * half][hn * 16 + n] = 0.0f;        // zero res
        }
    }

    // Per-wave max length (words are globally length-sorted; skip dead steps).
    int mwl = 0;
#pragma unroll
    for (int r = 0; r < 8; ++r) mwl = max(mwl, wl[r]);
    mwl = max(mwl, __shfl_xor(mwl, 16));
    mwl = max(mwl, __shfl_xor(mwl, 8));
    mwl = max(mwl, __shfl_xor(mwl, 4));
    mwl = max(mwl, __shfl_xor(mwl, 2));
    mwl = max(mwl, __shfl_xor(mwl, 1));

    const int wm = tile + n;  // A-matrix row (word) owned by this lane

    for (int t = 0; t < mwl; ++t) {
        // ---- A_x: embedding row of this word's char at step t (f32 -> bf16)
        int cid = chars[wm * MAX_WLEN + t];
        const float* er = emb + cid * EMB;
        bf16x16 ax0 = load_tile_f32(er, hi8);
        bf16x16 ax1 = load_tile_f32(er + 32, hi8);

        // ---- A_h: previous h from LDS staging (same-wave DS is in-order)
        const __bf16* hrow = &hstage[wave][n][0];
        bf16x16 ah0 = load_tile_bf16(hrow, 0, hi8);
        bf16x16 ah1 = load_tile_bf16(hrow, 32, hi8);
        bf16x16 ah2 = load_tile_bf16(hrow, 64, hi8);
        bf16x16 ah3 = load_tile_bf16(hrow, 96, hi8);

        float part[8];
#pragma unroll
        for (int r = 0; r < 8; ++r) part[r] = 0.0f;

        asm volatile("" ::: "memory");  // fence: keep B-tile loads per-iteration

        // ---- pass 1: gates + cell update + attention dot partials,
        //              16 hidden columns at a time
#pragma unroll
        for (int hn = 0; hn < 8; ++hn) {
            float aw = attn[hn * 16 + n];
            v8f acc[4];
#pragma unroll
            for (int g = 0; g < 4; ++g) {
                const int row = g * HID + hn * 16 + n;  // lane's B column = W row
                float bv = bsum[row];
                v8f a;
#pragma unroll
                for (int r = 0; r < 8; ++r) a[r] = bv;

                const __bf16* wr_i = wih + row * EMB;
                a = __builtin_amdgcn_wmma_f32_16x16x32_bf16(
                        false, ax0, false, load_tile_bf16(wr_i, 0, hi8), (short)0, a, false, false);
                a = __builtin_amdgcn_wmma_f32_16x16x32_bf16(
                        false, ax1, false, load_tile_bf16(wr_i, 32, hi8), (short)0, a, false, false);

                const __bf16* wr_h = whh + row * HID;
                a = __builtin_amdgcn_wmma_f32_16x16x32_bf16(
                        false, ah0, false, load_tile_bf16(wr_h, 0, hi8), (short)0, a, false, false);
                a = __builtin_amdgcn_wmma_f32_16x16x32_bf16(
                        false, ah1, false, load_tile_bf16(wr_h, 32, hi8), (short)0, a, false, false);
                a = __builtin_amdgcn_wmma_f32_16x16x32_bf16(
                        false, ah2, false, load_tile_bf16(wr_h, 64, hi8), (short)0, a, false, false);
                a = __builtin_amdgcn_wmma_f32_16x16x32_bf16(
                        false, ah3, false, load_tile_bf16(wr_h, 96, hi8), (short)0, a, false, false);
                acc[g] = a;
            }
#pragma unroll
            for (int r = 0; r < 8; ++r) {
                float iv = sigf(acc[0][r]);
                float fv = sigf(acc[1][r]);
                float gv = tanhf_fast(acc[2][r]);
                float ov = sigf(acc[3][r]);
                float cn = fv * creg[hn][r] + iv * gv;
                float hv = ov * tanhf_fast(cn);
                bool  m  = t < wl[r];
                creg[hn][r] = m ? cn : creg[hn][r];
                part[r] += (m ? hv : 0.0f) * aw;
                // Predicated store: finished words keep their old h in LDS.
                if (m) hstage[wave][r + 8 * half][hn * 16 + n] = (__bf16)hv;
            }
            asm volatile("" ::: "memory");  // fence between hn iterations
        }

        // ---- pass 2: finish dot across the 16 lanes of each half, then
        //              accumulate res += out * sigmoid(dot) from LDS h.
#pragma unroll
        for (int r = 0; r < 8; ++r) {
            float s = part[r];
            s += __shfl_xor(s, 1);
            s += __shfl_xor(s, 2);
            s += __shfl_xor(s, 4);
            s += __shfl_xor(s, 8);   // all 16 lanes of the half hold the dot
            float wt = sigf(s);
            if (t < wl[r]) {
#pragma unroll
                for (int hn = 0; hn < 8; ++hn) {
                    float ov = (float)hstage[wave][r + 8 * half][hn * 16 + n];
                    resbuf[wave][r + 8 * half][hn * 16 + n] += ov * wt;
                }
            }
        }
    }

    // ---- scatter through word_orig_idx: out[word_orig_idx[w]] = res[w]
#pragma unroll
    for (int r = 0; r < 8; ++r) {
        int w = tile + r + 8 * half;
        int o = word_orig_idx[w];
        float* op = out + o * HID + n;
#pragma unroll
        for (int hn = 0; hn < 8; ++hn)
            op[hn * 16] = resbuf[wave][r + 8 * half][hn * 16 + n];
    }
}

extern "C" void kernel_launch(void* const* d_in, const int* in_sizes, int n_in,
                              void* d_out, int out_size, void* d_ws, size_t ws_size,
                              hipStream_t stream) {
    const int*   chars         = (const int*)d_in[0];
    // d_in[1] = chars_mask (unused; derived from wordlens)
    const int*   word_orig_idx = (const int*)d_in[2];
    const int*   wordlens      = (const int*)d_in[3];
    // d_in[4] = sent_len scalar (output layout is flat-identical; unused)
    const float* emb           = (const float*)d_in[5];
    const float* w_ih          = (const float*)d_in[6];
    const float* w_hh          = (const float*)d_in[7];
    const float* b_ih          = (const float*)d_in[8];
    const float* b_hh          = (const float*)d_in[9];
    const float* attn          = (const float*)d_in[10];
    const float* h0            = (const float*)d_in[11];
    const float* c0            = (const float*)d_in[12];
    float*       out           = (float*)d_out;

    char* ws = (char*)d_ws;
    __bf16* wih_b = (__bf16*)(ws);                         // 512*64  bf16 = 64 KB
    __bf16* whh_b = (__bf16*)(ws + 64 * 1024);             // 512*128 bf16 = 128 KB
    float*  bsum  = (float*)(ws + 64 * 1024 + 128 * 1024); // 512 f32

    convert_weights_kernel<<<(512 * HID + 255) / 256, 256, 0, stream>>>(
        w_ih, w_hh, b_ih, b_hh, wih_b, whh_b, bsum);

    lstm_wmma_kernel<<<NUM_WORDS / WORDS_PER_BLOCK, WAVES_PER_BLOCK * 32, 0, stream>>>(
        chars, word_orig_idx, wordlens, emb, attn, h0, c0, wih_b, whh_b, bsum, out);
}
